// MultiHeadAttention_31799937860144
// MI455X (gfx1250) — matrix operand order repaired and checked by reading, not compiled, not run on hardware
//
#include <hip/hip_runtime.h>
#include <math.h>

// ---------------------------------------------------------------------------
// Types for WMMA (gfx1250, wave32)
// ---------------------------------------------------------------------------
typedef __bf16 bf16;
typedef __attribute__((ext_vector_type(8)))  bf16  v8bf;
typedef __attribute__((ext_vector_type(16))) bf16  v16bf;
typedef __attribute__((ext_vector_type(8)))  float v8f;
typedef int v4i __attribute__((vector_size(16)));   // matches builtin prototype

#define EMBED 1024
#define HEADS 16
#define HSZ   64
#define SEQ   2048
#define BATCH 4
#define MROWS (BATCH * SEQ)   // 8192

// ---------------------------------------------------------------------------
// CDNA5 async global->LDS copy (ASYNCcnt path), with safe fallback.
// ---------------------------------------------------------------------------
#if defined(__has_builtin)
#if __has_builtin(__builtin_amdgcn_global_load_async_to_lds_b128)
#define HAVE_ASYNC_LDS 1
#endif
#endif
#ifndef HAVE_ASYNC_LDS
#define HAVE_ASYNC_LDS 0
#endif

__device__ __forceinline__ void cp16_g2l(const bf16* gp, bf16* lp) {
#if HAVE_ASYNC_LDS
    const void* g = (const void*)gp;
    void*       l = (void*)lp;
    __builtin_amdgcn_global_load_async_to_lds_b128(
        (__attribute__((address_space(1))) v4i*)g,
        (__attribute__((address_space(3))) v4i*)l,
        0, 0);
#else
    *(v8bf*)lp = *(const v8bf*)gp;
#endif
}

__device__ __forceinline__ void wait_async0() {
#if HAVE_ASYNC_LDS
#if __has_builtin(__builtin_amdgcn_s_wait_asynccnt)
    __builtin_amdgcn_s_wait_asynccnt(0);
#else
    asm volatile("s_wait_asynccnt 0" ::: "memory");
#endif
#endif
}

// ---------------------------------------------------------------------------
// Fragment loaders (ISA 7.12.2 16-bit layouts, wave32)
//
// A-matrix 16x32 (MxK), 16-bit: lane L holds row M = L%16.
//   elements 0..7  = K = (L/16)*8 + 0..7        (contiguous 16B)
//   elements 8..15 = K = 16 + (L/16)*8 + 0..7   (contiguous 16B)
//
// B-matrix 32x16 (KxN), 16-bit: lane L holds column N = L%16.
//   elements 0..7  = K = (L/16)*8 + 0..7        (contiguous 16B)
//   elements 8..15 = K = 16 + (L/16)*8 + 0..7   (contiguous 16B)
// B sources are kept N-major (transposed weights / K rows / transposed V) so
// each lane reads contiguous memory.
// ---------------------------------------------------------------------------
__device__ __forceinline__ v16bf load_a_frag(const bf16* base, int ld, int lane) {
    const int row  = lane & 15;
    const int koff = (lane >> 4) << 3;          // 0 or 8
    const bf16* p0 = base + (size_t)row * ld + koff;
    const bf16* p1 = p0 + 16;
    v8bf lo = *(const v8bf*)p0;
    v8bf hi = *(const v8bf*)p1;
    return __builtin_shufflevector(lo, hi, 0,1,2,3,4,5,6,7,8,9,10,11,12,13,14,15);
}

__device__ __forceinline__ v16bf load_b_frag(const bf16* base, int ld, int lane) {
    const int n    = lane & 15;
    const int koff = (lane >> 4) << 3;          // 0 or 8
    const bf16* p  = base + (size_t)n * ld + koff;
    v8bf lo = *(const v8bf*)p;
    v8bf hi = *(const v8bf*)(p + 16);
    return __builtin_shufflevector(lo, hi, 0,1,2,3,4,5,6,7,8,9,10,11,12,13,14,15);
}

__device__ __forceinline__ v8f wmma_bf16(v16bf a, v16bf b, v8f c) {
    return __builtin_amdgcn_wmma_f32_16x16x32_bf16(false, a, false, b,
                                                   (short)0, c, false, false);
}

// ---------------------------------------------------------------------------
// Conversion kernels
// ---------------------------------------------------------------------------
__global__ void k_cvt_bf16(const float* __restrict__ src, bf16* __restrict__ dst, long n) {
    long i = (long)blockIdx.x * blockDim.x + threadIdx.x;
    if (i < n) dst[i] = (bf16)src[i];
}

// [H, E, hs] -> [H, hs, E]
__global__ void k_cvt_wT(const float* __restrict__ src, bf16* __restrict__ dst) {
    long i = (long)blockIdx.x * blockDim.x + threadIdx.x;  // over H*E*hs
    long n = (long)HEADS * EMBED * HSZ;
    if (i >= n) return;
    int h = (int)(i / (EMBED * HSZ));
    int r = (int)(i % (EMBED * HSZ));
    int e = r / HSZ;
    int d = r % HSZ;
    dst[((size_t)h * HSZ + d) * EMBED + e] = (bf16)src[i];
}

// [K=1024, N=1024] -> [N, K]
__global__ void k_cvt_woT(const float* __restrict__ src, bf16* __restrict__ dst) {
    long i = (long)blockIdx.x * blockDim.x + threadIdx.x;
    long n = (long)EMBED * EMBED;
    if (i >= n) return;
    int k = (int)(i / EMBED);
    int e = (int)(i % EMBED);
    dst[(size_t)e * EMBED + k] = (bf16)src[i];
}

// ---------------------------------------------------------------------------
// Stage one 64(N) x 64(K) bf16 B-tile into LDS (8 KB), 256 threads,
// 2 async b128 transfers per thread.  Row n is contiguous over K.
// ---------------------------------------------------------------------------
__device__ __forceinline__ void stage_btile(const bf16* __restrict__ Wt, int kb,
                                            bf16* lbuf, int tid) {
#pragma unroll
    for (int s2 = 0; s2 < 2; ++s2) {
        int seg = tid + s2 * 256;            // 0..511, 16B segments
        int n   = seg >> 3;                  // 0..63
        int ko  = (seg & 7) * 8;             // 0..56, bf16 elements
        cp16_g2l(Wt + (size_t)n * EMBED + kb + ko, lbuf + n * 64 + ko);
    }
}

// ---------------------------------------------------------------------------
// Fused QKV projection.  grid: (MROWS/128, HEADS, 3), block 256 (8 waves).
// Each wave: 16 rows x 64 cols; K-chunks of 64 double-buffered through LDS.
// Q,K stored [B,H,S,hs]; V stored transposed [B,H,hs,S].
// ---------------------------------------------------------------------------
__global__ __launch_bounds__(256) void k_qkv_gemm(
    const bf16* __restrict__ X,     // [MROWS, E]
    const bf16* __restrict__ WqT,   // [H, hs, E]
    const bf16* __restrict__ WkT,
    const bf16* __restrict__ WvT,
    bf16* __restrict__ Q,           // [B,H,S,hs]
    bf16* __restrict__ K,           // [B,H,S,hs]
    bf16* __restrict__ Vt)          // [B,H,hs,S]
{
    __shared__ __attribute__((aligned(16))) bf16 bstage[2][64 * 64];

    const int tid  = threadIdx.x;
    const int lane = tid & 31;
    const int wave = tid >> 5;
    const int h    = blockIdx.y;
    const int z    = blockIdx.z;    // 0=Q 1=K 2=V
    const int row0 = blockIdx.x * 128 + wave * 16;

    const bf16* Wt   = (z == 0 ? WqT : (z == 1 ? WkT : WvT)) + (size_t)h * HSZ * EMBED;
    const bf16* Arow = X + (size_t)row0 * EMBED;

    v8f acc[4] = {};
    stage_btile(Wt, 0, bstage[0], tid);

    for (int kb = 0, it = 0; kb < EMBED; kb += 64, ++it) {
        const int buf = it & 1;
        // per-wave A fragments (global, issued before the barrier)
        v16bf a0 = load_a_frag(Arow + kb,      EMBED, lane);
        v16bf a1 = load_a_frag(Arow + kb + 32, EMBED, lane);

        wait_async0();            // my async copies into bstage[buf] landed
        __syncthreads();          // everyone's copies landed; prev compute done
        if (kb + 64 < EMBED)
            stage_btile(Wt, kb + 64, bstage[buf ^ 1], tid);   // overlap w/ WMMA

        const bf16* lb = bstage[buf];
#pragma unroll
        for (int nt = 0; nt < 4; ++nt) {
            v16bf b0 = load_b_frag(lb + nt * 16 * 64,      64, lane);
            v16bf b1 = load_b_frag(lb + nt * 16 * 64 + 32, 64, lane);
            acc[nt] = wmma_bf16(a0, b0, acc[nt]);
            acc[nt] = wmma_bf16(a1, b1, acc[nt]);
        }
    }

    const int hi = lane >> 4;
    const int nn = lane & 15;
#pragma unroll
    for (int nt = 0; nt < 4; ++nt) {
#pragma unroll
        for (int r = 0; r < 8; ++r) {
            int gm = row0 + r + hi * 8;          // global row in [0, MROWS)
            int b_ = gm >> 11;                   // / SEQ
            int s  = gm & (SEQ - 1);
            int d  = nt * 16 + nn;
            bf16 v = (bf16)acc[nt][r];
            if (z == 0)
                Q[(((size_t)b_ * HEADS + h) * SEQ + s) * HSZ + d] = v;
            else if (z == 1)
                K[(((size_t)b_ * HEADS + h) * SEQ + s) * HSZ + d] = v;
            else
                Vt[(((size_t)b_ * HEADS + h) * HSZ + d) * SEQ + s] = v;
        }
    }
}

// ---------------------------------------------------------------------------
// Flash attention (causal, online softmax).
// grid: (SEQ/128, HEADS, BATCH), block 256 (8 waves).
// Each wave owns 16 query rows, sweeps keys in 64-wide tiles.
// ---------------------------------------------------------------------------
__global__ __launch_bounds__(256) void k_attn(
    const bf16* __restrict__ Q,    // [B,H,S,hs]
    const bf16* __restrict__ Km,   // [B,H,S,hs]
    const bf16* __restrict__ Vt,   // [B,H,hs,S]
    bf16* __restrict__ AO)         // [MROWS, H*hs]
{
    __shared__ __attribute__((aligned(16))) bf16 plds[8 * 16 * 64];

    const int lane = threadIdx.x & 31;
    const int wave = threadIdx.x >> 5;
    const int h    = blockIdx.y;
    const int b    = blockIdx.z;
    const int q0   = blockIdx.x * 128 + wave * 16;

    const bf16* Qbase = Q  + (((size_t)b * HEADS + h) * SEQ + q0) * HSZ;
    const bf16* Kbase = Km + (((size_t)b * HEADS + h) * SEQ) * HSZ;
    const bf16* Vbase = Vt + (((size_t)b * HEADS + h) * HSZ) * SEQ;
    bf16* myP = plds + wave * (16 * 64);

    // Q fragments (d split into two K=32 chunks), reused across all key tiles.
    v16bf qa0 = load_a_frag(Qbase, HSZ, lane);
    v16bf qa1 = load_a_frag(Qbase + 32, HSZ, lane);

    v8f   oacc[4] = {};
    float m_r[8], l_r[8];
#pragma unroll
    for (int r = 0; r < 8; ++r) { m_r[r] = -3.0e38f; l_r[r] = 0.0f; }

    const int   hi    = lane >> 4;
    const int   nn    = lane & 15;
    const float scale = 0.125f;    // hs^-0.5

    const int kend = q0 + 16;      // causal: keys < kend
    for (int k0 = 0; k0 < kend; k0 += 64) {
        // ---- scores S = Q K^T for 16 x 64 tile ----
        v8f sc[4] = {};
#pragma unroll
        for (int nt = 0; nt < 4; ++nt) {
            v16bf kf0 = load_b_frag(Kbase + (size_t)(k0 + nt * 16) * HSZ,      HSZ, lane);
            v16bf kf1 = load_b_frag(Kbase + (size_t)(k0 + nt * 16) * HSZ + 32, HSZ, lane);
            sc[nt] = wmma_bf16(qa0, kf0, sc[nt]);
            sc[nt] = wmma_bf16(qa1, kf1, sc[nt]);
        }

        // ---- scale + causal mask + online softmax update ----
#pragma unroll
        for (int r = 0; r < 8; ++r) {
            const int qidx = q0 + r + hi * 8;
            float mx = -3.0e38f;
#pragma unroll
            for (int nt = 0; nt < 4; ++nt) {
                float v = sc[nt][r] * scale;
                int kidx = k0 + nt * 16 + nn;
                v = (kidx <= qidx) ? v : -3.0e38f;
                sc[nt][r] = v;
                mx = fmaxf(mx, v);
            }
#pragma unroll
            for (int off = 1; off < 16; off <<= 1)
                mx = fmaxf(mx, __shfl_xor(mx, off, 32));

            const float mnew  = fmaxf(m_r[r], mx);
            const float alpha = __expf(m_r[r] - mnew);
            float rsum = 0.0f;
#pragma unroll
            for (int nt = 0; nt < 4; ++nt) {
                float p = __expf(sc[nt][r] - mnew);
                rsum += p;
                myP[(r + hi * 8) * 64 + nt * 16 + nn] = (bf16)p;
            }
#pragma unroll
            for (int off = 1; off < 16; off <<= 1)
                rsum += __shfl_xor(rsum, off, 32);

            l_r[r] = l_r[r] * alpha + rsum;
            m_r[r] = mnew;
#pragma unroll
            for (int nt = 0; nt < 4; ++nt)
                oacc[nt][r] *= alpha;
        }

        // P written by all lanes of this wave; LDS is in-order per wave, but
        // make the store->load ordering explicit.
        asm volatile("s_wait_dscnt 0" ::: "memory");

        // ---- O += P V  (P: 16x64 from LDS, V^T rows are contiguous) ----
#pragma unroll
        for (int kc = 0; kc < 2; ++kc) {
            v16bf pa = load_a_frag(myP + kc * 32, 64, lane);
#pragma unroll
            for (int nt = 0; nt < 4; ++nt) {
                v16bf vf = load_b_frag(Vbase + (size_t)(nt * 16) * SEQ + k0 + kc * 32,
                                       SEQ, lane);
                oacc[nt] = wmma_bf16(pa, vf, oacc[nt]);
            }
        }
        asm volatile("s_wait_dscnt 0" ::: "memory");
    }

    // ---- normalize and write out [MROWS, H*hs] in bf16 ----
#pragma unroll
    for (int r = 0; r < 8; ++r) {
        const float inv = 1.0f / l_r[r];
        const int qidx  = q0 + r + hi * 8;
        const size_t gm = (size_t)b * SEQ + qidx;
#pragma unroll
        for (int nt = 0; nt < 4; ++nt) {
            int col = h * HSZ + nt * 16 + nn;
            AO[gm * (HEADS * HSZ) + col] = (bf16)(oacc[nt][r] * inv);
        }
    }
}

// ---------------------------------------------------------------------------
// Output projection: out = AO @ Wo + bo   (fp32 out)
// grid: (MROWS/128, EMBED/64), block 256.  B-tile double-buffered via LDS.
// ---------------------------------------------------------------------------
__global__ __launch_bounds__(256) void k_out_proj(
    const bf16* __restrict__ AO,    // [MROWS, 1024]
    const bf16* __restrict__ WoT,   // [1024(eout), 1024(k)]
    const float* __restrict__ bias, // [1024]
    float* __restrict__ out)        // [MROWS, 1024]
{
    __shared__ __attribute__((aligned(16))) bf16 bstage[2][64 * 64];

    const int tid  = threadIdx.x;
    const int lane = tid & 31;
    const int wave = tid >> 5;
    const int row0 = blockIdx.x * 128 + wave * 16;
    const int n0   = blockIdx.y * 64;

    const bf16* Wt   = WoT + (size_t)n0 * EMBED;
    const bf16* Arow = AO + (size_t)row0 * EMBED;

    v8f acc[4] = {};
    stage_btile(Wt, 0, bstage[0], tid);

    for (int kb = 0, it = 0; kb < EMBED; kb += 64, ++it) {
        const int buf = it & 1;
        v16bf a0 = load_a_frag(Arow + kb,      EMBED, lane);
        v16bf a1 = load_a_frag(Arow + kb + 32, EMBED, lane);

        wait_async0();
        __syncthreads();
        if (kb + 64 < EMBED)
            stage_btile(Wt, kb + 64, bstage[buf ^ 1], tid);

        const bf16* lb = bstage[buf];
#pragma unroll
        for (int nt = 0; nt < 4; ++nt) {
            v16bf b0 = load_b_frag(lb + nt * 16 * 64,      64, lane);
            v16bf b1 = load_b_frag(lb + nt * 16 * 64 + 32, 64, lane);
            acc[nt] = wmma_bf16(a0, b0, acc[nt]);
            acc[nt] = wmma_bf16(a1, b1, acc[nt]);
        }
    }

    const int hi = lane >> 4;
    const int nn = lane & 15;
#pragma unroll
    for (int nt = 0; nt < 4; ++nt) {
        const int col = n0 + nt * 16 + nn;
        const float bv = bias[col];
#pragma unroll
        for (int r = 0; r < 8; ++r) {
            const size_t gm = (size_t)row0 + r + hi * 8;
            out[gm * EMBED + col] = acc[nt][r] + bv;
        }
    }
}

// ---------------------------------------------------------------------------
// Host-side launcher
// ---------------------------------------------------------------------------
extern "C" void kernel_launch(void* const* d_in, const int* in_sizes, int n_in,
                              void* d_out, int out_size, void* d_ws, size_t ws_size,
                              hipStream_t stream) {
    const float* x  = (const float*)d_in[0];
    const float* Wq = (const float*)d_in[1];
    const float* Wk = (const float*)d_in[2];
    const float* Wv = (const float*)d_in[3];
    const float* Wo = (const float*)d_in[4];
    const float* bo = (const float*)d_in[5];
    float* out = (float*)d_out;

    char* w = (char*)d_ws;
    bf16* xb  = (bf16*)w;  w += (size_t)MROWS * EMBED * 2;                 // 16.8 MB
    bf16* wqt = (bf16*)w;  w += (size_t)HEADS * HSZ * EMBED * 2;          //  2.1 MB
    bf16* wkt = (bf16*)w;  w += (size_t)HEADS * HSZ * EMBED * 2;
    bf16* wvt = (bf16*)w;  w += (size_t)HEADS * HSZ * EMBED * 2;
    bf16* wot = (bf16*)w;  w += (size_t)EMBED * EMBED * 2;                //  2.1 MB
    bf16* Qb  = (bf16*)w;  w += (size_t)BATCH * HEADS * SEQ * HSZ * 2;    // 16.8 MB
    bf16* Kb  = (bf16*)w;  w += (size_t)BATCH * HEADS * SEQ * HSZ * 2;
    bf16* Vtb = (bf16*)w;  w += (size_t)BATCH * HEADS * SEQ * HSZ * 2;
    bf16* AO  = (bf16*)w;  w += (size_t)MROWS * EMBED * 2;                // 16.8 MB

    // 1) precision / layout conversion
    {
        long n = (long)MROWS * EMBED;
        k_cvt_bf16<<<dim3((unsigned)((n + 255) / 256)), dim3(256), 0, stream>>>(x, xb, n);
        long nw = (long)HEADS * EMBED * HSZ;
        unsigned gw = (unsigned)((nw + 255) / 256);
        k_cvt_wT<<<dim3(gw), dim3(256), 0, stream>>>(Wq, wqt);
        k_cvt_wT<<<dim3(gw), dim3(256), 0, stream>>>(Wk, wkt);
        k_cvt_wT<<<dim3(gw), dim3(256), 0, stream>>>(Wv, wvt);
        long no = (long)EMBED * EMBED;
        k_cvt_woT<<<dim3((unsigned)((no + 255) / 256)), dim3(256), 0, stream>>>(Wo, wot);
    }

    // 2) fused QKV projection (WMMA bf16, async-LDS staged B)
    k_qkv_gemm<<<dim3(MROWS / 128, HEADS, 3), dim3(256), 0, stream>>>(
        xb, wqt, wkt, wvt, Qb, Kb, Vtb);

    // 3) causal flash attention (WMMA bf16)
    k_attn<<<dim3(SEQ / 128, HEADS, BATCH), dim3(256), 0, stream>>>(Qb, Kb, Vtb, AO);

    // 4) output projection + bias (WMMA bf16, async-LDS staged B, fp32 out)
    k_out_proj<<<dim3(MROWS / 128, EMBED / 64), dim3(256), 0, stream>>>(AO, wot, bo, out);
}
